// Perpendicular_DeformableConv2d_69063074119904
// MI455X (gfx1250) — matrix operand order repaired and checked by reading, not compiled, not run on hardware
//
#include <hip/hip_runtime.h>
#include <hip/hip_bf16.h>

typedef __attribute__((ext_vector_type(16))) _Float16 v16h;
typedef __attribute__((ext_vector_type(8)))  float    v8f;

#define C_IN   64
#define O_CH   64
#define HH     96
#define WW     96
#define BB     4
#define RAD    5
#define WIN    11                    // 11x11 integer-offset window
#define NOFF   (WIN*WIN)             // 121
#define KTOT   (NOFF*C_IN)           // 7744 == 242 * 32
#define TILE_W 32
#define COLS   (TILE_W + 2*RAD)      // 42
#define ROWS   WIN                   // 11
#define CST    C_IN                  // c-stride inside LDS tile (keeps ds_load_b128 16B aligned)

union AF { v16h v; uint4 q[2]; };

// ---------------------------------------------------------------------------
// Kernel 1: fold the constant bilinear corner weights of each deformable tap
// into an effective 11x11 conv kernel, stored fp16 as w_eff[o][off*64 + c].
// ---------------------------------------------------------------------------
__global__ __launch_bounds__(256) void k_build_weff(
    const float* __restrict__ frames,   // (O, 18)
    const float* __restrict__ convw,    // (O, C, 3, 3)
    _Float16* __restrict__ weff)        // (O, 7744)
{
    __shared__ float acc[KTOT];         // 121*64 fp32 = 31 KB
    const int o = blockIdx.x;

    for (int i = threadIdx.x; i < KTOT; i += 256) acc[i] = 0.f;
    __syncthreads();

    if (threadIdx.x < C_IN) {
        const int c = threadIdx.x;
        for (int n = 0; n < 9; ++n) {
            float px = frames[o*18 + n];
            float py = frames[o*18 + 9 + n];
            float fx = floorf(px), fy = floorf(py);
            float ax = px - fx,    ay = py - fy;
            int   ix = (int)fx,    iy = (int)fy;
            float wv = convw[((o*C_IN + c)*3 + n/3)*3 + (n%3)];
            const float gx[2] = {1.f - ax, ax};
            const float gy[2] = {1.f - ay, ay};
            for (int qy = 0; qy < 2; ++qy)
                for (int qx = 0; qx < 2; ++qx) {
                    int dx = min(max(ix + qx, -RAD), RAD);
                    int dy = min(max(iy + qy, -RAD), RAD);
                    int off = (dy + RAD)*WIN + (dx + RAD);
                    acc[off*C_IN + c] += gx[qx]*gy[qy]*wv;
                }
        }
    }
    __syncthreads();

    for (int i = threadIdx.x; i < KTOT; i += 256)
        weff[(size_t)o*KTOT + i] = (_Float16)acc[i];
}

// ---------------------------------------------------------------------------
// Kernel 2: implicit-GEMM.  Block tile = 32 pixels (one (b,h) row segment) x
// 64 channels.  8 wave32s: pg in {0,1} pixel groups x 4 o-groups of 16.
// K walk = nested (row, d) counted loops so every ds/global access is an
// immediate offset from a per-row base (no div/mod SALU chains), and two
// independent WMMA accumulator chains hide the f16 WMMA RAW latency.
// ---------------------------------------------------------------------------
__global__ __launch_bounds__(256) void k_deform_gemm(
    const float* __restrict__ x,        // (B, C, H, W)
    const _Float16* __restrict__ weff,  // (O, 7744)
    float* __restrict__ out)            // (B, O, H, W)  (pre-BN)
{
    __shared__ _Float16 xt[ROWS*COLS*CST];   // 11*42*64 fp16 = 57.8 KB

    const int blk = blockIdx.x;              // b*H*3 + h*3 + tw
    const int tw  = blk % 3;
    const int bh  = blk / 3;
    const int h   = bh % HH;
    const int b   = bh / HH;
    const int w0  = tw * TILE_W;

    // stage x tile with replicate padding, fp32 -> fp16 (coalesced over col)
    for (int i = threadIdx.x; i < C_IN*ROWS*COLS; i += 256) {
        int col = i % COLS;
        int t   = i / COLS;
        int row = t % ROWS;
        int c   = t / ROWS;
        int hh  = min(max(h  - RAD + row, 0), HH-1);
        int ww  = min(max(w0 - RAD + col, 0), WW-1);
        xt[(row*COLS + col)*CST + c] = (_Float16)x[((b*C_IN + c)*HH + hh)*WW + ww];
    }
    __syncthreads();

    const int lane  = threadIdx.x & 31;
    const int wave  = threadIdx.x >> 5;
    const int pg    = wave & 1;
    const int obase = (wave >> 1) * 16;
    const int hi    = lane >> 4;             // K-half selector
    const int ln    = lane & 15;
    const int a_kb  = hi ? 16 : 0;           // A: lanes>=16 take K {16..31}

    v8f acc0 = {};                           // chain for channel-half 0
    v8f acc1 = {};                           // chain for channel-half 1

    // lane-invariant bases
    const _Float16* arow = &xt[(pg*16 + ln)*CST] + a_kb;          // + row*COLS*CST + d*CST
    const _Float16* wrow = weff + (size_t)(obase + ln)*KTOT + (hi << 4);  // + (row*WIN+d)*C_IN

    #pragma unroll 1
    for (int row = 0; row < ROWS; ++row) {
        const _Float16* ar = arow + row*(COLS*CST);
        const _Float16* wr = wrow + row*(WIN*C_IN);
        #pragma unroll
        for (int d = 0; d < WIN; ++d) {
            const _Float16* ab = ar + d*CST;      // immediate ds offsets
            const _Float16* bb = wr + d*C_IN;     // immediate global offsets
            __builtin_prefetch(bb + C_IN, 0, 1);  // global_prefetch_b8: next tap's weights

            AF a0, a1, b0, b1;
            a0.q[0] = *(const uint4*)(ab);
            a0.q[1] = *(const uint4*)(ab + 8);
            b0.q[0] = *(const uint4*)(bb);
            b0.q[1] = *(const uint4*)(bb + 8);
            a1.q[0] = *(const uint4*)(ab + 32);
            a1.q[1] = *(const uint4*)(ab + 40);
            b1.q[0] = *(const uint4*)(bb + 32);
            b1.q[1] = *(const uint4*)(bb + 40);

            acc0 = __builtin_amdgcn_wmma_f32_16x16x32_f16(
                       false, a0.v, false, b0.v, (short)0, acc0, false, false);
            acc1 = __builtin_amdgcn_wmma_f32_16x16x32_f16(
                       false, a1.v, false, b1.v, (short)0, acc1, false, false);
        }
    }

    const v8f acc = acc0 + acc1;

    // D layout: lane ln = o column, vgpr j -> pixel M = j + 8*hi  => 8
    // consecutive w per lane: two aligned float4 stores.
    const int o  = obase + ln;
    const int wq = w0 + pg*16 + 8*hi;
    float* op = &out[(((size_t)b*O_CH + o)*HH + h)*WW + wq];
    *(float4*)(op)     = make_float4(acc[0], acc[1], acc[2], acc[3]);
    *(float4*)(op + 4) = make_float4(acc[4], acc[5], acc[6], acc[7]);
}

// ---------------------------------------------------------------------------
// Kernel 3: per-channel batch statistics (biased variance).
// Note conv_b cancels in (out - mean), so it was never added.
// ---------------------------------------------------------------------------
__global__ __launch_bounds__(256) void k_bn_stats(
    const float* __restrict__ out, float* __restrict__ mean, float* __restrict__ istd)
{
    __shared__ float s1[256], s2[256];
    const int o = blockIdx.x;
    float sa = 0.f, sq = 0.f;
    for (int bi = 0; bi < BB; ++bi) {
        const float* p = out + ((size_t)(bi*O_CH + o))*HH*WW;
        for (int i = threadIdx.x; i < HH*WW; i += 256) {
            float v = p[i]; sa += v; sq += v*v;
        }
    }
    s1[threadIdx.x] = sa; s2[threadIdx.x] = sq;
    __syncthreads();
    for (int st = 128; st > 0; st >>= 1) {
        if (threadIdx.x < st) {
            s1[threadIdx.x] += s1[threadIdx.x + st];
            s2[threadIdx.x] += s2[threadIdx.x + st];
        }
        __syncthreads();
    }
    if (threadIdx.x == 0) {
        const float n = (float)(BB*HH*WW);
        float m = s1[0] / n;
        float v = s2[0] / n - m*m;
        mean[o] = m;
        istd[o] = rsqrtf(v + 1e-5f);
    }
}

// ---------------------------------------------------------------------------
// Kernel 4: in-place normalize + affine + ReLU.
// ---------------------------------------------------------------------------
__global__ __launch_bounds__(256) void k_bn_apply(
    float* __restrict__ out,
    const float* __restrict__ mean, const float* __restrict__ istd,
    const float* __restrict__ gamma, const float* __restrict__ beta)
{
    int idx = blockIdx.x * 256 + threadIdx.x;
    if (idx >= BB*O_CH*HH*WW) return;
    int o = (idx / (HH*WW)) & (O_CH - 1);
    float v = out[idx];
    v = gamma[o] * (v - mean[o]) * istd[o] + beta[o];
    out[idx] = fmaxf(v, 0.f);
}

// ---------------------------------------------------------------------------
extern "C" void kernel_launch(void* const* d_in, const int* in_sizes, int n_in,
                              void* d_out, int out_size, void* d_ws, size_t ws_size,
                              hipStream_t stream) {
    const float* x      = (const float*)d_in[0];
    const float* frames = (const float*)d_in[1];
    const float* convw  = (const float*)d_in[2];
    // d_in[3] = conv_b: a per-channel constant, cancelled by train-mode BN.
    const float* gamma  = (const float*)d_in[4];
    const float* beta   = (const float*)d_in[5];
    float* out = (float*)d_out;

    _Float16* weff = (_Float16*)d_ws;                           // 64*7744*2 = 968 KB
    float* mean = (float*)((char*)d_ws + (size_t)O_CH*KTOT*sizeof(_Float16));
    float* istd = mean + O_CH;

    k_build_weff<<<O_CH, 256, 0, stream>>>(frames, convw, weff);
    k_deform_gemm<<<BB*HH*(WW/TILE_W), 256, 0, stream>>>(x, weff, out);
    k_bn_stats<<<O_CH, 256, 0, stream>>>(out, mean, istd);
    const int tot = BB*O_CH*HH*WW;
    k_bn_apply<<<(tot + 255)/256, 256, 0, stream>>>(out, mean, istd, gamma, beta);
}
